// MaskedSelfAttention_88124138979519
// MI455X (gfx1250) — matrix operand order repaired and checked by reading, not compiled, not run on hardware
//
#include <hip/hip_runtime.h>
#include <hip/hip_bf16.h>
#include <math.h>

// CDNA5 / gfx1250: wave32, WMMA 16x16x32 bf16 with fp32 accumulation.
typedef __attribute__((ext_vector_type(16))) __bf16       v16bf;
typedef __attribute__((ext_vector_type(8)))  float        v8f;
typedef __attribute__((ext_vector_type(4)))  unsigned int v4u;

union Frag { v16bf bf; v4u u[2]; };

#define SEQ 4096
#define DM  2048

#define BM 128      // block tile M (2 waves x 64)
#define BN 256      // block tile N (4 waves x 64)
#define BK 32       // K step per WMMA sweep
#define PITCH 40    // LDS row pitch in halves (32 + 8 pad -> 80B, 16B aligned)

// NT GEMM: C[M,N] = A[M,K] (row-major, lda) * B[N,K]^T (row-major, ldb)
// 256 threads = 8 waves, wave grid 2(M) x 4(N), 64x64 wave tile = 4x4 WMMA tiles.
// OUT_BF16: store bf16, else fp32.
// CAUSAL:   epilogue writes -1e30 for col>row; fully-masked blocks skip compute.
// SOFTMAX_A: A holds raw scores (bf16); fragment elements become
//            exp(s - rowmax) * inv_rowsum before the matmul; K-loop clipped
//            at the causal diagonal of this row block.
template<bool OUT_BF16, bool CAUSAL, bool SOFTMAX_A>
__global__ __launch_bounds__(256)
void gemm_nt_bf16(const __bf16* __restrict__ A, const __bf16* __restrict__ B,
                  void* __restrict__ Cv, const float* __restrict__ stats,
                  int M, int N, int K, int lda, int ldb, int ldc, float alpha)
{
    __shared__ __bf16 smA[2][BM * PITCH];   // 2 x 5120 halves
    __shared__ __bf16 smB[2][BN * PITCH];   // 2 x 10240 halves

    const int tid   = threadIdx.x;
    const int lane  = tid & 31;
    const int wid   = tid >> 5;      // 0..7
    const int waveM = wid & 1;       // 0..1 -> 64 rows each
    const int waveN = wid >> 1;      // 0..3 -> 64 cols each

    const int rowBase = blockIdx.y * BM;
    const int colBase = blockIdx.x * BN;

    const bool skipCompute = CAUSAL && (colBase >= rowBase + BM);

    int kEnd = K;
    if (SOFTMAX_A) { int lim = rowBase + BM; kEnd = (lim < K) ? lim : K; }
    const int KT = kEnd / BK;

    v8f acc[4][4] = {};

    // Per-lane softmax stats for the four A-fragment rows this lane owns.
    float smx[4] = {0.f, 0.f, 0.f, 0.f};
    float sil[4] = {1.f, 1.f, 1.f, 1.f};
    if (SOFTMAX_A) {
        #pragma unroll
        for (int tm = 0; tm < 4; ++tm) {
            const int r = rowBase + waveM * 64 + tm * 16 + (lane & 15);
            smx[tm] = stats[2 * r];
            sil[tm] = stats[2 * r + 1];
        }
    }

    if (!skipCompute) {
        v4u ra[2], rb[4];
        // prologue: stage K-tile 0 (A: 512 16B chunks, B: 1024 16B chunks)
        #pragma unroll
        for (int i = 0; i < 2; ++i) {
            const int c = tid + i * 256, row = c >> 2, q = c & 3;
            ra[i] = *(const v4u*)(A + (size_t)(rowBase + row) * lda + q * 8);
        }
        #pragma unroll
        for (int i = 0; i < 4; ++i) {
            const int c = tid + i * 256, row = c >> 2, q = c & 3;
            rb[i] = *(const v4u*)(B + (size_t)(colBase + row) * ldb + q * 8);
        }
        #pragma unroll
        for (int i = 0; i < 2; ++i) {
            const int c = tid + i * 256, row = c >> 2, q = c & 3;
            *(v4u*)&smA[0][row * PITCH + q * 8] = ra[i];
        }
        #pragma unroll
        for (int i = 0; i < 4; ++i) {
            const int c = tid + i * 256, row = c >> 2, q = c & 3;
            *(v4u*)&smB[0][row * PITCH + q * 8] = rb[i];
        }
        __syncthreads();

        for (int kt = 0; kt < KT; ++kt) {
            const int  cur  = kt & 1;
            const bool more = (kt + 1) < KT;
            if (more) {
                const int k0 = (kt + 1) * BK;
                #pragma unroll
                for (int i = 0; i < 2; ++i) {
                    const int c = tid + i * 256, row = c >> 2, q = c & 3;
                    ra[i] = *(const v4u*)(A + (size_t)(rowBase + row) * lda + k0 + q * 8);
                }
                #pragma unroll
                for (int i = 0; i < 4; ++i) {
                    const int c = tid + i * 256, row = c >> 2, q = c & 3;
                    rb[i] = *(const v4u*)(B + (size_t)(colBase + row) * ldb + k0 + q * 8);
                }
                if (kt + 2 < KT) {  // L2 prefetch of the tile after next
                    const int kp = (kt + 2) * BK;
                    __builtin_prefetch(A + (size_t)(rowBase + (tid >> 1)) * lda + kp, 0, 3);
                    __builtin_prefetch(B + (size_t)(colBase + tid) * ldb + kp, 0, 3);
                }
            }

            // A fragments: 16-bit A 16x32 layout — lane L holds row M=L%16,
            // halves 0..7 = K kb..kb+7, halves 8..15 = K kb+16..kb+23, kb=(L>=16)*8.
            Frag afr[4], bfr[4];
            const int arow = waveM * 64 + (lane & 15);
            const int kbA  = (lane >> 4) * 8;
            #pragma unroll
            for (int tm = 0; tm < 4; ++tm) {
                const __bf16* p = &smA[cur][(arow + tm * 16) * PITCH + kbA];
                afr[tm].u[0] = *(const v4u*)p;
                afr[tm].u[1] = *(const v4u*)(p + 16);
                if (SOFTMAX_A) {
                    #pragma unroll
                    for (int e = 0; e < 16; ++e) {
                        const float s = (float)afr[tm].bf[e];
                        afr[tm].bf[e] = (__bf16)(__expf(s - smx[tm]) * sil[tm]);
                    }
                }
            }
            // B fragments: 32x16 B layout — lane L holds column N=L%16,
            // halves 0..7 = K kb..kb+7, halves 8..15 = K kb+16..kb+23, kb=(L>=16)*8.
            const int brow = waveN * 64 + (lane & 15);
            const int kbB  = (lane >> 4) * 8;
            #pragma unroll
            for (int tn = 0; tn < 4; ++tn) {
                const __bf16* p = &smB[cur][(brow + tn * 16) * PITCH + kbB];
                bfr[tn].u[0] = *(const v4u*)p;
                bfr[tn].u[1] = *(const v4u*)(p + 16);
            }

            #pragma unroll
            for (int tm = 0; tm < 4; ++tm)
                #pragma unroll
                for (int tn = 0; tn < 4; ++tn)
                    acc[tm][tn] = __builtin_amdgcn_wmma_f32_16x16x32_bf16(
                        false, afr[tm].bf, false, bfr[tn].bf,
                        (short)0, acc[tm][tn], false, false);

            __syncthreads();
            if (more) {
                #pragma unroll
                for (int i = 0; i < 2; ++i) {
                    const int c = tid + i * 256, row = c >> 2, q = c & 3;
                    *(v4u*)&smA[cur ^ 1][row * PITCH + q * 8] = ra[i];
                }
                #pragma unroll
                for (int i = 0; i < 4; ++i) {
                    const int c = tid + i * 256, row = c >> 2, q = c & 3;
                    *(v4u*)&smB[cur ^ 1][row * PITCH + q * 8] = rb[i];
                }
                __syncthreads();
            }
        }
    }

    // Epilogue. C/D layout: lane L holds col N=L%16; VGPR r holds row r (+8 if L>=16).
    __bf16* Cb = (__bf16*)Cv;
    float*  Cf = (float*)Cv;
    const int mhalf = (lane >> 4) * 8;
    const int ncol  = lane & 15;
    #pragma unroll
    for (int tm = 0; tm < 4; ++tm)
        #pragma unroll
        for (int tn = 0; tn < 4; ++tn)
            #pragma unroll
            for (int r = 0; r < 8; ++r) {
                const int grow = rowBase + waveM * 64 + tm * 16 + mhalf + r;
                const int gcol = colBase + waveN * 64 + tn * 16 + ncol;
                float v = acc[tm][tn][r] * alpha;
                if (CAUSAL && (gcol > grow)) v = -1e30f;
                const size_t idx = (size_t)grow * ldc + gcol;
                if (OUT_BF16) Cb[idx] = (__bf16)v;
                else          Cf[idx] = v;
            }
}

__global__ void f32_to_bf16(const float* __restrict__ src,
                            __bf16* __restrict__ dst, int n)
{
    int i = blockIdx.x * blockDim.x + threadIdx.x;
    const int stride = gridDim.x * blockDim.x;
    for (; i < n; i += stride) dst[i] = (__bf16)src[i];
}

// Per-row softmax stats over the stored (masked) score matrix:
// stats[2r] = rowmax, stats[2r+1] = 1/sum(exp(s - rowmax)).
__global__ __launch_bounds__(256)
void row_softmax_stats(const __bf16* __restrict__ S, float* __restrict__ stats, int n)
{
    __shared__ float red[256];
    const int row = blockIdx.x;
    const int tid = threadIdx.x;
    const __bf16* p = S + (size_t)row * n;

    float mx = -3.0e38f;
    for (int j = tid; j < n; j += 256) mx = fmaxf(mx, (float)p[j]);
    red[tid] = mx; __syncthreads();
    for (int s = 128; s > 0; s >>= 1) {
        if (tid < s) red[tid] = fmaxf(red[tid], red[tid + s]);
        __syncthreads();
    }
    mx = red[0]; __syncthreads();

    float sum = 0.f;
    for (int j = tid; j < n; j += 256) sum += __expf((float)p[j] - mx);
    red[tid] = sum; __syncthreads();
    for (int s = 128; s > 0; s >>= 1) {
        if (tid < s) red[tid] += red[tid + s];
        __syncthreads();
    }
    if (tid == 0) { stats[2 * row] = mx; stats[2 * row + 1] = 1.0f / red[0]; }
}

extern "C" void kernel_launch(void* const* d_in, const int* in_sizes, int n_in,
                              void* d_out, int out_size, void* d_ws, size_t ws_size,
                              hipStream_t stream)
{
    (void)in_sizes; (void)n_in; (void)out_size; (void)ws_size;
    const float* X  = (const float*)d_in[0];
    // d_in[1] = mask (bool, strictly-upper-triangular) -> handled analytically.
    const float* Wq = (const float*)d_in[2];
    const float* Wk = (const float*)d_in[3];
    const float* Wv = (const float*)d_in[4];

    char* ws = (char*)d_ws;
    size_t off = 0;
    auto alloc = [&](size_t bytes) -> void* {
        void* p = ws + off; off += (bytes + 255) & ~(size_t)255; return p;
    };
    __bf16* XB  = (__bf16*)alloc((size_t)SEQ * DM * 2);   // X bf16        [S,D]
    __bf16* WqB = (__bf16*)alloc((size_t)DM * DM * 2);
    __bf16* WkB = (__bf16*)alloc((size_t)DM * DM * 2);
    __bf16* WvB = (__bf16*)alloc((size_t)DM * DM * 2);
    __bf16* Qb  = (__bf16*)alloc((size_t)SEQ * DM * 2);   // Q*1/sqrt(D)   [S,D]
    __bf16* Kb  = (__bf16*)alloc((size_t)SEQ * DM * 2);   // K             [S,D]
    __bf16* VT  = (__bf16*)alloc((size_t)DM * SEQ * 2);   // V^T           [D,S]
    __bf16* Sb  = (__bf16*)alloc((size_t)SEQ * SEQ * 2);  // masked scores [S,S]
    float*  st  = (float*)alloc((size_t)SEQ * 2 * 4);     // (max, 1/sum) per row

    f32_to_bf16<<<512, 256, 0, stream>>>(X,  XB,  SEQ * DM);
    f32_to_bf16<<<512, 256, 0, stream>>>(Wq, WqB, DM * DM);
    f32_to_bf16<<<512, 256, 0, stream>>>(Wk, WkB, DM * DM);
    f32_to_bf16<<<512, 256, 0, stream>>>(Wv, WvB, DM * DM);

    const float qscale = 1.0f / sqrtf((float)DM);
    const dim3 blk(256);

    // Q = X*Wq^T * (1/sqrt(D))   [S,D]
    gemm_nt_bf16<true, false, false><<<dim3(DM / BN, SEQ / BM), blk, 0, stream>>>(
        XB, WqB, Qb, nullptr, SEQ, DM, DM, DM, DM, DM, qscale);
    // K = X*Wk^T                 [S,D]
    gemm_nt_bf16<true, false, false><<<dim3(DM / BN, SEQ / BM), blk, 0, stream>>>(
        XB, WkB, Kb, nullptr, SEQ, DM, DM, DM, DM, DM, 1.0f);
    // V^T = Wv*X^T               [D,S]  (V computed pre-transposed: P@V stays NT)
    gemm_nt_bf16<true, false, false><<<dim3(SEQ / BN, DM / BM), blk, 0, stream>>>(
        WvB, XB, VT, nullptr, DM, SEQ, DM, DM, DM, SEQ, 1.0f);
    // S = Q*K^T with causal mask [S,S]; above-diagonal blocks skip compute
    gemm_nt_bf16<true, true, false><<<dim3(SEQ / BN, SEQ / BM), blk, 0, stream>>>(
        Qb, Kb, Sb, nullptr, SEQ, SEQ, DM, DM, DM, SEQ, 1.0f);
    // row softmax statistics
    row_softmax_stats<<<SEQ, 256, 0, stream>>>(Sb, st, SEQ);
    // out = softmax(S) @ V  (A-fragments normalized on the fly; K clipped at diagonal)
    gemm_nt_bf16<false, false, true><<<dim3(DM / BN, SEQ / BM), blk, 0, stream>>>(
        Sb, VT, d_out, st, SEQ, DM, SEQ, SEQ, SEQ, DM, 1.0f);
}
